// RNNWithLocalAttention_87832081203398
// MI455X (gfx1250) — matrix operand order repaired and checked by reading, not compiled, not run on hardware
//
#include <hip/hip_runtime.h>
#include <hip/hip_bf16.h>

// ---------------------------------------------------------------------------
// RNN + local attention for MI455X (gfx1250, wave32, WMMA bf16)
//
//   1. embed gather (fp32 wte -> bf16 x)
//   2. xg = x @ w_ih^T + b_ih + b_hh               (GEMM, fp32 out)
//   3. LSTM recurrence: 16 persistent WGs, global spin barrier per step,
//      h @ w_hh^T via M-padded (2->16) bf16 WMMA; h broadcast reload via TDM
//   4. q = h Wq^T, k = h Wk^T, vT = (h Wv^T)^T     (GEMMs, bf16 out)
//   5. scores = q k^T / sqrt(H)  (batched GEMM, fp32)
//   6. masked softmax (j >= i-16) -> bf16 att
//   7. y = att @ v  (batched GEMM via vT, bf16 out)
//   8. o = y Wo^T + bo + h (residual)              (GEMM, fp32 out)
//   9. layernorm -> bf16
//  10. logits = ln @ W_head^T + b_head             (GEMM, fp32 -> d_out)
//
// GEMM: one wave owns a 32(M) x 64(N) strip (8 accumulators), K pipelined in
// 64-element chunks with double-buffered fragment registers so global_load_b128
// overlaps v_wmma issue instead of draining loadcnt before every WMMA.
// ---------------------------------------------------------------------------

typedef unsigned short u16;
typedef __bf16 v16bf __attribute__((ext_vector_type(16)));
typedef float  v8f   __attribute__((ext_vector_type(8)));
typedef unsigned int v4u __attribute__((ext_vector_type(4)));
typedef int v4i __attribute__((ext_vector_type(4)));
typedef int v8i __attribute__((ext_vector_type(8)));

union FragBF {
  v16bf v;
  uint4 q[2];
};

__device__ __forceinline__ u16 f2bf(float f) {
  unsigned u = __float_as_uint(f);
  unsigned r = (u + 0x7FFFu + ((u >> 16) & 1u)) >> 16;
  return (u16)r;
}
__device__ __forceinline__ v8f v8f_zero() {
  v8f z;
#pragma unroll
  for (int i = 0; i < 8; ++i) z[i] = 0.0f;
  return z;
}
__device__ __forceinline__ float sigm(float x) { return 1.0f / (1.0f + __expf(-x)); }

#define WMMA_BF16(A, B, C) \
  __builtin_amdgcn_wmma_f32_16x16x32_bf16(false, (A).v, false, (B).v, (short)0, (C), false, false)

// ---------------------------------------------------------------------------
// Elementwise converters
// ---------------------------------------------------------------------------
__global__ void cvt_kernel(const float* __restrict__ in, u16* __restrict__ out, long long n) {
  long long i = (long long)blockIdx.x * blockDim.x + threadIdx.x;
  long long stride = (long long)gridDim.x * blockDim.x;
  for (; i < n; i += stride) out[i] = f2bf(in[i]);
}

__global__ void embed_kernel(const int* __restrict__ idx, const float* __restrict__ wte,
                             u16* __restrict__ x) {
  int row = blockIdx.x;                       // row in [0, B*T)
  int tok = idx[row];
  const float* w = wte + (long long)tok * 1024;
  u16* o = x + (long long)row * 1024;
  for (int c = threadIdx.x; c < 1024; c += blockDim.x) o[c] = f2bf(w[c]);
}

// ---------------------------------------------------------------------------
// Generic bf16 WMMA GEMM:  C[m,n] = scale * sum_k A[m,k]*W[n,k] + bias(+bias2)
//   (+ residual), optional batch, optional transposed bf16 store (for v^T).
// Requirements: M%32==0, N%64==0, K%64==0.
// Fragments: A 16x32 / B 32x16 bf16 with contiguous k (k = 16*hi + s), f32 C 16x16 per CDNA5 ISA 7.12.2.
// ---------------------------------------------------------------------------
struct GemmP {
  const u16* A; const u16* W;
  const float* bias; const float* bias2; const float* residual;
  float* outF; u16* outH;
  long long sA, sW, sO;   // per-batch element strides
  int lda, ldw, ldo, ldr;
  int M, N, K;            // per-batch sizes
  float scale;
  int transOut;           // bf16 out stored [n*ldo + m]
};

// Load one k-step (32 K-elements) of fragments: 2 A rows-of-16, 4 B tiles.
#define GEMM_LOAD(A0, A1, B0, B1, B2, B3, KK)                                   \
  do {                                                                          \
    const uint4* ap0 = (const uint4*)(aRow0 + (KK));                            \
    (A0).q[0] = ap0[0]; (A0).q[1] = ap0[1];                                     \
    const uint4* ap1 = (const uint4*)(aRow1 + (KK));                            \
    (A1).q[0] = ap1[0]; (A1).q[1] = ap1[1];                                     \
    const uint4* bp0 = (const uint4*)(wRow0 + (KK));                            \
    (B0).q[0] = bp0[0]; (B0).q[1] = bp0[1];                                     \
    const uint4* bp1 = (const uint4*)(wRow1 + (KK));                            \
    (B1).q[0] = bp1[0]; (B1).q[1] = bp1[1];                                     \
    const uint4* bp2 = (const uint4*)(wRow2 + (KK));                            \
    (B2).q[0] = bp2[0]; (B2).q[1] = bp2[1];                                     \
    const uint4* bp3 = (const uint4*)(wRow3 + (KK));                            \
    (B3).q[0] = bp3[0]; (B3).q[1] = bp3[1];                                     \
  } while (0)

#define GEMM_MMA(A0, A1, B0, B1, B2, B3)                                        \
  do {                                                                          \
    c00 = WMMA_BF16(A0, B0, c00); c01 = WMMA_BF16(A0, B1, c01);                 \
    c02 = WMMA_BF16(A0, B2, c02); c03 = WMMA_BF16(A0, B3, c03);                 \
    c10 = WMMA_BF16(A1, B0, c10); c11 = WMMA_BF16(A1, B1, c11);                 \
    c12 = WMMA_BF16(A1, B2, c12); c13 = WMMA_BF16(A1, B3, c13);                 \
  } while (0)

__global__ __launch_bounds__(256) void gemm_bf16_kernel(GemmP P) {
  const int tid  = threadIdx.x;
  const int lane = tid & 31;
  const int wid  = tid >> 5;
  const int hi   = lane >> 4;   // half-wave: 0 or 1
  const int l16  = lane & 15;

  const int stripsN = P.N >> 6;
  const int tilesM  = P.M >> 5;
  const int wg = blockIdx.x * 8 + wid;
  if (wg >= tilesM * stripsN) return;
  const int mTile  = wg / stripsN;
  const int nStrip = wg % stripsN;
  const long long b = blockIdx.y;

  const u16* A = P.A + b * P.sA;
  const u16* W = P.W + b * P.sW;
  const int mBase = mTile << 5;
  const int nBase = nStrip << 6;

  v8f c00 = v8f_zero(), c01 = v8f_zero(), c02 = v8f_zero(), c03 = v8f_zero();
  v8f c10 = v8f_zero(), c11 = v8f_zero(), c12 = v8f_zero(), c13 = v8f_zero();

  // A fragment rows: 32B contiguous at k0 + 16*hi.
  const u16* aRow0 = A + (long long)(mBase + l16) * P.lda + 16 * hi;
  const u16* aRow1 = A + (long long)(mBase + 16 + l16) * P.lda + 16 * hi;
  // B fragment rows: 32B contiguous at k0 + 16*hi.
  const u16* wRow0 = W + (long long)(nBase +  0 + l16) * P.ldw + 16 * hi;
  const u16* wRow1 = W + (long long)(nBase + 16 + l16) * P.ldw + 16 * hi;
  const u16* wRow2 = W + (long long)(nBase + 32 + l16) * P.ldw + 16 * hi;
  const u16* wRow3 = W + (long long)(nBase + 48 + l16) * P.ldw + 16 * hi;

  FragBF xa0, xa1, xb0, xb1, xb2, xb3;
  FragBF ya0, ya1, yb0, yb1, yb2, yb3;

  GEMM_LOAD(xa0, xa1, xb0, xb1, xb2, xb3, 0);
  for (int k0 = 0; k0 < P.K; k0 += 64) {
    GEMM_LOAD(ya0, ya1, yb0, yb1, yb2, yb3, k0 + 32);   // K%64==0 -> always valid
    GEMM_MMA(xa0, xa1, xb0, xb1, xb2, xb3);
    if (k0 + 64 < P.K) GEMM_LOAD(xa0, xa1, xb0, xb1, xb2, xb3, k0 + 64);
    GEMM_MMA(ya0, ya1, yb0, yb1, yb2, yb3);
  }

  v8f accs[2][4] = {{c00, c01, c02, c03}, {c10, c11, c12, c13}};
#pragma unroll
  for (int j = 0; j < 4; ++j) {
    const int n = nBase + j * 16 + l16;
    float bv = 0.0f;
    if (P.bias)  bv += P.bias[n];
    if (P.bias2) bv += P.bias2[n];
#pragma unroll
    for (int half = 0; half < 2; ++half) {
#pragma unroll
      for (int r = 0; r < 8; ++r) {
        const int m = mBase + half * 16 + r + 8 * hi;  // C layout: VGPR r = row r+8*hi
        float v = accs[half][j][r] * P.scale + bv;
        if (P.residual) v += P.residual[(long long)m * P.ldr + n];
        if (P.outF) {
          P.outF[b * P.sO + (long long)m * P.ldo + n] = v;
        } else if (P.transOut) {
          P.outH[b * P.sO + (long long)n * P.ldo + m] = f2bf(v);
        } else {
          P.outH[b * P.sO + (long long)m * P.ldo + n] = f2bf(v);
        }
      }
    }
  }
}

// ---------------------------------------------------------------------------
// LSTM recurrence. 16 persistent workgroups x 256 threads (8 waves).
// WG w owns hidden units [w*64, w*64+64): gate columns {g*1024 + u}, g=0..3
// => 256 columns = 16 WMMA tiles => 2 tiles per wave.
// A matrix = current h [2,1024] bf16 padded to 16 rows (rows 2..15 = 0).
// Cross-WG sync: release/acquire atomic counter + s_sleep spin.
// h broadcast reload (global h_shared -> LDS) done by the Tensor Data Mover.
// ---------------------------------------------------------------------------
#define LSTM_LOAD(AF, B0, B1, KK)                                               \
  do {                                                                          \
    if (l16 < 2) {                                                              \
      const uint4* ap = (const uint4*)(&h_lds[l16][(KK) + 16 * hi]);            \
      (AF).q[0] = ap[0]; (AF).q[1] = ap[1];                                     \
    } else {                                                                    \
      (AF).q[0] = make_uint4(0u, 0u, 0u, 0u);                                   \
      (AF).q[1] = make_uint4(0u, 0u, 0u, 0u);                                   \
    }                                                                           \
    const uint4* bp0 = (const uint4*)(whh0 + (KK));                             \
    (B0).q[0] = bp0[0]; (B0).q[1] = bp0[1];                                     \
    const uint4* bp1 = (const uint4*)(whh1 + (KK));                             \
    (B1).q[0] = bp1[0]; (B1).q[1] = bp1[1];                                     \
  } while (0)

__global__ __launch_bounds__(256) void lstm_kernel(const float* __restrict__ xg,
                                                   const u16* __restrict__ whh,
                                                   float* __restrict__ h_out,
                                                   u16* __restrict__ h_bf_all,
                                                   u16* __restrict__ h_shared,
                                                   unsigned int* __restrict__ cnt) {
  __shared__ __align__(16) u16 h_lds[2][1024];   // current h, bf16
  __shared__ float c_lds[2][64];                 // this WG's cell state
  __shared__ float g_lds[2][256];                // this WG's gate pre-activations

  const int tid  = threadIdx.x;
  const int lane = tid & 31;
  const int wid  = tid >> 5;
  const int hi   = lane >> 4;
  const int l16  = lane & 15;
  const int u0   = blockIdx.x * 64;

  for (int i = tid; i < 2048; i += 256) ((u16*)h_lds)[i] = 0;
  if (tid < 128) c_lds[tid >> 6][tid & 63] = 0.0f;
  __syncthreads();

  const int tile0 = wid * 2, tile1 = wid * 2 + 1;
  const int n0 = (tile0 >> 2) * 1024 + u0 + (tile0 & 3) * 16;  // global gate col base
  const int n1 = (tile1 >> 2) * 1024 + u0 + (tile1 & 3) * 16;
  const u16* whh0 = whh + (long long)(n0 + l16) * 1024 + 16 * hi;
  const u16* whh1 = whh + (long long)(n1 + l16) * 1024 + 16 * hi;

#if __has_builtin(__builtin_amdgcn_tensor_load_to_lds) && __has_builtin(__builtin_amdgcn_s_wait_tensorcnt)
  // Tensor DMA descriptor: 1-D tile of 2048 x 2B from h_shared -> h_lds.
  // Group0: count=1; lds_addr; global_addr[56:0] at bits 120:64; type=2.
  // Group1: data_size=1 (2B) at bits 17:16; tensor_dim0=2048 at bits 79:48;
  //         tensor_dim1=1 at bits 111:80; tile_dim0=2048 at bits 127:112;
  //         tensor_dim0_stride=2048 at bits 207:160.
  union { v4u v; unsigned w[4]; } g0;
  union { v8i v; unsigned w[8]; } g1;
  {
    const unsigned long long ga = (unsigned long long)h_shared;
    const unsigned ldsOff = (unsigned)(unsigned long long)(&h_lds[0][0]);
    g0.w[0] = 1u;
    g0.w[1] = ldsOff;
    g0.w[2] = (unsigned)ga;
    g0.w[3] = (unsigned)((ga >> 32) & 0x01FFFFFFull) | (2u << 30);
    g1.w[0] = (1u << 16);
    g1.w[1] = (2048u & 0xFFFFu) << 16;
    g1.w[2] = (2048u >> 16) | (1u << 16);
    g1.w[3] = (2048u << 16);
    g1.w[4] = 0u;
    g1.w[5] = 2048u;
    g1.w[6] = 0u;
    g1.w[7] = 0u;
  }
  v4i z4; for (int i = 0; i < 4; ++i) z4[i] = 0;
#if defined(__clang_major__) && (__clang_major__ >= 23)
  v8i z8; for (int i = 0; i < 8; ++i) z8[i] = 0;
#endif
#endif

  for (int t = 0; t < 2048; ++t) {
    v8f a0 = v8f_zero(), a1 = v8f_zero();
    {
      FragBF xa, xb0, xb1, ya, yb0, yb1;
      LSTM_LOAD(xa, xb0, xb1, 0);
      for (int k0 = 0; k0 < 1024; k0 += 64) {
        LSTM_LOAD(ya, yb0, yb1, k0 + 32);
        a0 = WMMA_BF16(xa, xb0, a0);
        a1 = WMMA_BF16(xa, xb1, a1);
        if (k0 + 64 < 1024) LSTM_LOAD(xa, xb0, xb1, k0 + 64);
        a0 = WMMA_BF16(ya, yb0, a0);
        a1 = WMMA_BF16(ya, yb1, a1);
      }
    }
    if (hi == 0) {
      // C layout: lanes 0-15, VGPR 0 = row M=0 (batch 0), VGPR 1 = M=1 (batch 1)
      const long long x0 = (long long)t * 4096;             // (0*T+t)*4H
      const long long x1 = (2048LL + t) * 4096;             // batch 1
      g_lds[0][tile0 * 16 + l16] = a0[0] + xg[x0 + n0 + l16];
      g_lds[1][tile0 * 16 + l16] = a0[1] + xg[x1 + n0 + l16];
      g_lds[0][tile1 * 16 + l16] = a1[0] + xg[x0 + n1 + l16];
      g_lds[1][tile1 * 16 + l16] = a1[1] + xg[x1 + n1 + l16];
      if (t + 1 < 2048) {   // warm caches for next step's xg slice
        __builtin_prefetch(&xg[x0 + 4096 + n0 + l16], 0, 1);
        __builtin_prefetch(&xg[x1 + 4096 + n0 + l16], 0, 1);
      }
    }
    __syncthreads();

    if (tid < 128) {
      const int b = tid >> 6, ul = tid & 63, u = u0 + ul;
      const float ig = sigm(g_lds[b][ul]);
      const float fg = sigm(g_lds[b][64 + ul]);
      const float gg = tanhf(g_lds[b][128 + ul]);
      const float og = sigm(g_lds[b][192 + ul]);
      const float c = fg * c_lds[b][ul] + ig * gg;
      c_lds[b][ul] = c;
      const float h = og * tanhf(c);
      const long long o = ((long long)b * 2048 + t) * 1024 + u;
      h_out[o] = h;
      const u16 hb = f2bf(h);
      h_bf_all[o] = hb;
      h_shared[b * 1024 + u] = hb;
    }
    __threadfence();
    __syncthreads();
    if (tid == 0) {
      __hip_atomic_fetch_add(cnt, 1u, __ATOMIC_ACQ_REL, __HIP_MEMORY_SCOPE_AGENT);
      const unsigned target = 16u * (unsigned)(t + 1);
      while (__hip_atomic_load(cnt, __ATOMIC_ACQUIRE, __HIP_MEMORY_SCOPE_AGENT) < target)
        __builtin_amdgcn_s_sleep(1);
    }
    __syncthreads();
    // Reload full h[2,1024] into LDS for the next step.
#if __has_builtin(__builtin_amdgcn_tensor_load_to_lds) && __has_builtin(__builtin_amdgcn_s_wait_tensorcnt)
    if (wid == 0) {
#if defined(__clang_major__) && (__clang_major__ >= 23)
      __builtin_amdgcn_tensor_load_to_lds(g0.v, g1.v, z4, z4, z8, 0);
#else
      __builtin_amdgcn_tensor_load_to_lds(g0.v, g1.v, z4, z4, 0);
#endif
      __builtin_amdgcn_s_wait_tensorcnt(0);
    }
    __syncthreads();
#else
    for (int i = tid; i < 2048; i += 256) ((u16*)h_lds)[i] = h_shared[i];
    __syncthreads();
#endif
  }
}

// ---------------------------------------------------------------------------
// Masked softmax: row i over j in [0,2048), allowed iff j >= i-16.
// ---------------------------------------------------------------------------
__global__ __launch_bounds__(256) void softmax_kernel(const float* __restrict__ s,
                                                      u16* __restrict__ att) {
  const int lane = threadIdx.x & 31;
  const int wid  = threadIdx.x >> 5;
  const int i = blockIdx.x * 8 + wid;
  const long long b = blockIdx.y;
  const float* row = s + ((b * 2048 + i) * 2048LL);
  u16* orow = att + ((b * 2048 + i) * 2048LL);
  int jmin = i - 16; if (jmin < 0) jmin = 0;

  float mx = -3.0e38f;
  for (int j = lane; j < 2048; j += 32)
    if (j >= jmin) mx = fmaxf(mx, row[j]);
#pragma unroll
  for (int m = 16; m >= 1; m >>= 1) mx = fmaxf(mx, __shfl_xor(mx, m, 32));

  float sum = 0.0f;
  for (int j = lane; j < 2048; j += 32)
    if (j >= jmin) sum += __expf(row[j] - mx);
#pragma unroll
  for (int m = 16; m >= 1; m >>= 1) sum += __shfl_xor(sum, m, 32);
  const float inv = 1.0f / sum;

  for (int j = lane; j < 2048; j += 32)
    orow[j] = (j >= jmin) ? f2bf(__expf(row[j] - mx) * inv) : (u16)0;
}

// ---------------------------------------------------------------------------
// LayerNorm over H=1024, writes bf16 for the head GEMM.
// ---------------------------------------------------------------------------
__global__ __launch_bounds__(256) void layernorm_kernel(const float* __restrict__ x,
                                                        const float* __restrict__ gam,
                                                        const float* __restrict__ bet,
                                                        u16* __restrict__ out) {
  const int row = blockIdx.x;
  const int tid = threadIdx.x;
  const float* xr = x + (long long)row * 1024;
  float vals[4];
  float s = 0.0f, s2 = 0.0f;
#pragma unroll
  for (int i = 0; i < 4; ++i) {
    const float v = xr[tid * 4 + i];
    vals[i] = v; s += v; s2 += v * v;
  }
#pragma unroll
  for (int m = 16; m >= 1; m >>= 1) {
    s  += __shfl_xor(s, m, 32);
    s2 += __shfl_xor(s2, m, 32);
  }
  __shared__ float w1[8], w2[8], stat[2];
  if ((tid & 31) == 0) { w1[tid >> 5] = s; w2[tid >> 5] = s2; }
  __syncthreads();
  if (tid == 0) {
    float t1 = 0.0f, t2 = 0.0f;
    for (int i = 0; i < 8; ++i) { t1 += w1[i]; t2 += w2[i]; }
    const float mu = t1 * (1.0f / 1024.0f);
    const float var = t2 * (1.0f / 1024.0f) - mu * mu;
    stat[0] = mu;
    stat[1] = rsqrtf(var + 1e-5f);
  }
  __syncthreads();
  const float mu = stat[0], rstd = stat[1];
  u16* orow = out + (long long)row * 1024;
#pragma unroll
  for (int i = 0; i < 4; ++i) {
    const int c = tid * 4 + i;
    orow[c] = f2bf((vals[i] - mu) * rstd * gam[c] + bet[c]);
  }
}

// ---------------------------------------------------------------------------
// Host side
// ---------------------------------------------------------------------------
static void launch_gemm(hipStream_t s, const GemmP& P, int batch) {
  const int tiles = (P.M / 32) * (P.N / 64);
  dim3 grid((tiles + 7) / 8, batch);
  hipLaunchKernelGGL(gemm_bf16_kernel, grid, dim3(256), 0, s, P);
}

static void launch_cvt(hipStream_t s, const float* in, u16* out, long long n) {
  long long blocks = (n + 1023) / 1024;
  if (blocks > 16384) blocks = 16384;
  hipLaunchKernelGGL(cvt_kernel, dim3((unsigned)blocks), dim3(256), 0, s, in, out, n);
}

extern "C" void kernel_launch(void* const* d_in, const int* in_sizes, int n_in,
                              void* d_out, int out_size, void* d_ws, size_t ws_size,
                              hipStream_t stream) {
  (void)in_sizes; (void)n_in; (void)out_size; (void)ws_size;
  const int*   idx    = (const int*)d_in[0];
  const float* wte    = (const float*)d_in[1];
  const float* w_ih   = (const float*)d_in[2];
  const float* w_hh   = (const float*)d_in[3];
  const float* b_ih   = (const float*)d_in[4];
  const float* b_hh   = (const float*)d_in[5];
  const float* Wq     = (const float*)d_in[6];
  const float* bq     = (const float*)d_in[7];
  const float* Wk     = (const float*)d_in[8];
  const float* bk     = (const float*)d_in[9];
  const float* Wv     = (const float*)d_in[10];
  const float* bv     = (const float*)d_in[11];
  const float* Wo     = (const float*)d_in[12];
  const float* bo     = (const float*)d_in[13];
  const float* ln_g   = (const float*)d_in[14];
  const float* ln_b   = (const float*)d_in[15];
  const float* W_head = (const float*)d_in[16];
  const float* b_head = (const float*)d_in[17];
  float* logits = (float*)d_out;

  const long long BT = 4096, H = 1024, T = 2048, V = 32000;

  // ---- workspace carve (256B aligned) ----
  char* base = (char*)d_ws;
  size_t off = 0;
  auto carve = [&](size_t bytes) -> void* {
    void* r = base + off;
    off = (off + bytes + 255) & ~(size_t)255;
    return r;
  };
  u16*  x_bf     = (u16*)carve(BT * H * 2);
  u16*  wih_bf   = (u16*)carve(4096LL * 1024 * 2);
  u16*  whh_bf   = (u16*)carve(4096LL * 1024 * 2);
  u16*  wq_bf    = (u16*)carve(H * H * 2);
  u16*  wk_bf    = (u16*)carve(H * H * 2);
  u16*  wv_bf    = (u16*)carve(H * H * 2);
  u16*  wo_bf    = (u16*)carve(H * H * 2);
  u16*  whead_bf = (u16*)carve(V * H * 2);
  float* xg      = (float*)carve(BT * 4096LL * 4);
  float* h_out   = (float*)carve(BT * H * 4);
  u16*  h_bf     = (u16*)carve(BT * H * 2);
  u16*  h_shared = (u16*)carve(2 * H * 2);
  u16*  q_bf     = (u16*)carve(BT * H * 2);
  u16*  k_bf     = (u16*)carve(BT * H * 2);
  u16*  vT_bf    = (u16*)carve(BT * H * 2);
  float* scores  = (float*)carve(2LL * T * T * 4);
  u16*  att_bf   = (u16*)carve(2LL * T * T * 2);
  u16*  y_bf     = (u16*)carve(BT * H * 2);
  float* xres    = (float*)carve(BT * H * 4);
  u16*  ln_bf    = (u16*)carve(BT * H * 2);
  unsigned int* cnt = (unsigned int*)carve(64);

  hipMemsetAsync(cnt, 0, 64, stream);

  // ---- weight conversions to bf16 ----
  launch_cvt(stream, w_ih,   wih_bf,   4096LL * 1024);
  launch_cvt(stream, w_hh,   whh_bf,   4096LL * 1024);
  launch_cvt(stream, Wq,     wq_bf,    H * H);
  launch_cvt(stream, Wk,     wk_bf,    H * H);
  launch_cvt(stream, Wv,     wv_bf,    H * H);
  launch_cvt(stream, Wo,     wo_bf,    H * H);
  launch_cvt(stream, W_head, whead_bf, V * H);

  // ---- 1. embedding gather -> bf16 ----
  hipLaunchKernelGGL(embed_kernel, dim3((unsigned)BT), dim3(256), 0, stream, idx, wte, x_bf);

  // ---- 2. xg = x @ w_ih^T + b_ih + b_hh ----
  {
    GemmP P{};
    P.A = x_bf; P.W = wih_bf; P.bias = b_ih; P.bias2 = b_hh;
    P.outF = xg; P.lda = 1024; P.ldw = 1024; P.ldo = 4096;
    P.M = 4096; P.N = 4096; P.K = 1024; P.scale = 1.0f;
    launch_gemm(stream, P, 1);
  }

  // ---- 3. LSTM recurrence (16 persistent WGs) ----
  hipLaunchKernelGGL(lstm_kernel, dim3(16), dim3(256), 0, stream,
                     xg, whh_bf, h_out, h_bf, h_shared, cnt);

  // ---- 4. q, k, vT projections ----
  {
    GemmP P{};
    P.A = h_bf; P.W = wq_bf; P.bias = bq; P.outH = q_bf;
    P.lda = 1024; P.ldw = 1024; P.ldo = 1024;
    P.M = 4096; P.N = 1024; P.K = 1024; P.scale = 1.0f;
    launch_gemm(stream, P, 1);
    P.W = wk_bf; P.bias = bk; P.outH = k_bf;
    launch_gemm(stream, P, 1);
  }
  {
    GemmP P{};   // vT[b][c][t] = v[b][t][c], batched with transposed store
    P.A = h_bf; P.sA = 2048LL * 1024; P.lda = 1024;
    P.W = wv_bf; P.sW = 0; P.ldw = 1024;
    P.bias = bv;
    P.outH = vT_bf; P.sO = 1024LL * 2048; P.ldo = 2048; P.transOut = 1;
    P.M = 2048; P.N = 1024; P.K = 1024; P.scale = 1.0f;
    launch_gemm(stream, P, 2);
  }

  // ---- 5. scores = q k^T / sqrt(H) ----
  {
    GemmP P{};
    P.A = q_bf; P.sA = 2048LL * 1024; P.lda = 1024;
    P.W = k_bf; P.sW = 2048LL * 1024; P.ldw = 1024;
    P.outF = scores; P.sO = 2048LL * 2048; P.ldo = 2048;
    P.M = 2048; P.N = 2048; P.K = 1024; P.scale = 0.03125f;  // 1/sqrt(1024)
    launch_gemm(stream, P, 2);
  }

  // ---- 6. masked softmax -> bf16 attention ----
  hipLaunchKernelGGL(softmax_kernel, dim3(256, 2), dim3(256), 0, stream, scores, att_bf);

  // ---- 7. y = att @ v  (via vT as the [N,K] weight) ----
  {
    GemmP P{};
    P.A = att_bf; P.sA = 2048LL * 2048; P.lda = 2048;
    P.W = vT_bf;  P.sW = 1024LL * 2048; P.ldw = 2048;
    P.outH = y_bf; P.sO = 2048LL * 1024; P.ldo = 1024;
    P.M = 2048; P.N = 1024; P.K = 2048; P.scale = 1.0f;
    launch_gemm(stream, P, 2);
  }

  // ---- 8. o = y Wo^T + bo + h (residual) ----
  {
    GemmP P{};
    P.A = y_bf; P.W = wo_bf; P.bias = bo; P.residual = h_out; P.ldr = 1024;
    P.outF = xres; P.lda = 1024; P.ldw = 1024; P.ldo = 1024;
    P.M = 4096; P.N = 1024; P.K = 1024; P.scale = 1.0f;
    launch_gemm(stream, P, 1);
  }

  // ---- 9. layernorm -> bf16 ----
  hipLaunchKernelGGL(layernorm_kernel, dim3((unsigned)BT), dim3(256), 0, stream,
                     xres, ln_g, ln_b, ln_bf);

  // ---- 10. logits = ln @ W_head^T + b_head ----
  {
    GemmP P{};
    P.A = ln_bf; P.W = whead_bf; P.bias = b_head;
    P.outF = logits; P.lda = 1024; P.ldw = 1024; P.ldo = 32000;
    P.M = 4096; P.N = 32000; P.K = 1024; P.scale = 1.0f;
    launch_gemm(stream, P, 1);
  }
}
